// VisualTransformer_42142219108896
// MI455X (gfx1250) — matrix operand order repaired and checked by reading, not compiled, not run on hardware
//
#include <hip/hip_runtime.h>
#include <cstdint>
#include <cstddef>

// ---------------------------------------------------------------------------
// CDNA5 (gfx1250, wave32) WMMA types
// ---------------------------------------------------------------------------
typedef __bf16 bf16_t;
typedef bf16_t v16bf __attribute__((ext_vector_type(16)));
typedef float  v8f   __attribute__((ext_vector_type(8)));

union FragU { v16bf v; uint4 q[2]; };

#define NB    16      // batch
#define CFM   1024
#define HWL   3136    // real pixel count
#define HWP   3200    // padded to 25*128 so M/K tiles are always exact
#define CTOK  1024
#define NTOK  128
#define TOKCN (CTOK * NTOK)   // 131072 elements per batch of token tensors

// ---------------------------------------------------------------------------
// NT GEMM:  C(MxN) = alpha * A(MxK,row-major) * B(NxK,row-major)^T  [+R][relu]
// BM=BN=128, BK=64, 256 threads = 8 waves in a 4(M) x 2(N) grid; each wave
// computes a 32x64 tile = 2x4 accumulators -> 16 v_wmma per k-stage.
//
// Staging uses the CDNA5 async copy engine: GLOBAL_LOAD_ASYNC_TO_LDS_B128
// writes tiles straight into a double-buffered LDS region (no staging VGPRs,
// so the kernel fits the 128-VGPR budget of a 256-thread block without
// spills). Tile k+1's async loads are issued before computing tile k;
// in-order ASYNCcnt completion means `s_wait_asynccnt 0x8` retires exactly
// the 8 loads of the current tile while the next 8 stay in flight.
// Host guarantees: M % 128 == 0, K % 64 == 0, all staged rows addressable.
// ---------------------------------------------------------------------------
#define GEMM_BK      64
#define GEMM_LDS_LD  72                       // 64 + 8 pad bf16 -> 144B rows
#define TILE_ELEMS   (128 * GEMM_LDS_LD)      // 9216 bf16 per tile
#define TILE_BYTES   (TILE_ELEMS * 2)         // 18432 B
#define GEMM_LDS_BYTES (4 * TILE_BYTES)       // A0 B0 A1 B1 = 73728 B dynamic

__global__ __launch_bounds__(256) void gemm_nt_wmma(
    const bf16_t* __restrict__ A,  long long sA, int lda,
    const bf16_t* __restrict__ Bm, long long sB, int ldb,
    float alpha,
    const float*  __restrict__ Rz, long long sR,
    float*        __restrict__ Fo, long long sF,
    bf16_t*       __restrict__ Bo, long long sBo,
    int M, int N, int K, int ldc, int relu)
{
  constexpr int BK = GEMM_BK, LDS_LD = GEMM_LDS_LD;
  extern __shared__ __align__(16) char smem[];   // [A0][B0][A1][B1]
  const int tid  = threadIdx.x;
  const int wave = tid >> 5;
  const int lane = tid & 31;
  const int mw   = wave & 3;        // 4 M-groups of 32 rows
  const int nw   = wave >> 2;       // 2 N-groups of 64 cols
  const int lm   = lane & 15;
  const int lhi  = (lane >> 4) & 1; // lane half selects K-subgroup
  const int batch = blockIdx.z;
  A  += (long long)batch * sA;
  Bm += (long long)batch * sB;
  const float* Rp = Rz ? Rz + (long long)batch * sR  : nullptr;
  float*       Fp = Fo ? Fo + (long long)batch * sF  : nullptr;
  bf16_t*      Bp = Bo ? Bo + (long long)batch * sBo : nullptr;
  const int m0 = blockIdx.y * 128, n0 = blockIdx.x * 128;

  // Per-thread staging geometry: 8 bf16 per chunk, 4 A + 4 B chunks/stage.
  const int r0 = tid >> 3;          // 0..31, +32*i covers 128 rows
  const int kk = (tid & 7) * 8;     // 0..56
  const bf16_t* Ab = A  + (size_t)(m0 + r0) * lda + kk;
  const bf16_t* Bb = Bm + (size_t)(n0 + r0) * ldb + kk;
  // LDS dest addresses: flat LDS-aperture pointers carry the LDS byte
  // offset in addr[31:0] (ISA 10.2), which is what VDST expects.
  const unsigned ldsA0 =
      (unsigned)(uintptr_t)(smem + (size_t)(r0 * LDS_LD + kk) * 2);

  // Issue one tile's 8 async b128 copies (global -> LDS buffer b).
  auto issue_tile = [&](int kcol, int b) {
    const unsigned abase = ldsA0 + (unsigned)(b * 2 * TILE_BYTES);
#pragma unroll
    for (int i = 0; i < 4; ++i) {
      unsigned la = abase + (unsigned)(i * 32 * LDS_LD * 2);
      unsigned lb = la + (unsigned)TILE_BYTES;
      unsigned long long ga =
          (unsigned long long)(uintptr_t)(Ab + (size_t)(32 * i) * lda + kcol);
      unsigned long long gb =
          (unsigned long long)(uintptr_t)(Bb + (size_t)(32 * i) * ldb + kcol);
      asm volatile("global_load_async_to_lds_b128 %0, %1, off"
                   :: "v"(la), "v"(ga) : "memory");
      asm volatile("global_load_async_to_lds_b128 %0, %1, off"
                   :: "v"(lb), "v"(gb) : "memory");
    }
  };

  v8f acc[2][4];
#pragma unroll
  for (int am = 0; am < 2; ++am)
#pragma unroll
    for (int bn = 0; bn < 4; ++bn) acc[am][bn] = (v8f){0,0,0,0,0,0,0,0};

  issue_tile(0, 0);                  // prologue: tile 0 -> buffer 0
  int buf = 0;

  for (int k0 = 0; k0 < K; k0 += BK) {
    // Issue tile k+1 into the other buffer (its readers finished at the
    // previous iteration's trailing barrier). Last iter reloads k=0.
    const int kn = (k0 + BK < K) ? (k0 + BK) : 0;
    issue_tile(kn, buf ^ 1);
    // In-order ASYNCcnt: <=8 outstanding leaves only the 8 just issued.
    asm volatile("s_wait_asynccnt 0x8" ::: "memory");
    __syncthreads();                 // all waves' tile-k data visible

    const bf16_t* AsB = (const bf16_t*)smem + (size_t)buf * 2 * TILE_ELEMS;
    const bf16_t* BsB = AsB + TILE_ELEMS;

    // ---- compute: two 16x16x32 K-substeps, 16 WMMA total ----
#pragma unroll
    for (int ks = 0; ks < 2; ++ks) {
      const int kb = ks * 32;
      // A frags: ISA 16-bit A 16x32 layout (lanes 0-15: e0-7->K0-7,
      // e8-15->K16-23; lanes 16-31: +8)
      FragU a[2];
#pragma unroll
      for (int am = 0; am < 2; ++am) {
        int ar  = mw * 32 + am * 16 + lm;
        int ako = kb + lhi * 8;
        a[am].q[0] = *(const uint4*)(AsB + ar * LDS_LD + ako);
        a[am].q[1] = *(const uint4*)(AsB + ar * LDS_LD + ako + 16);
      }
      // B frags: 32x16 layout (N = lane%16, same K order as A)
      FragU bq[4];
#pragma unroll
      for (int bn = 0; bn < 4; ++bn) {
        int br  = nw * 64 + bn * 16 + lm;
        int bko = kb + lhi * 8;
        bq[bn].q[0] = *(const uint4*)(BsB + br * LDS_LD + bko);
        bq[bn].q[1] = *(const uint4*)(BsB + br * LDS_LD + bko + 16);
      }
#pragma unroll
      for (int am = 0; am < 2; ++am)
#pragma unroll
        for (int bn = 0; bn < 4; ++bn)
          acc[am][bn] = __builtin_amdgcn_wmma_f32_16x16x32_bf16(
              false, a[am].v, false, bq[bn].v, (short)0, acc[am][bn],
              false, false);
    }
    __syncthreads();                 // readers done: buffer reusable
    buf ^= 1;
  }
  // Final redundant tile's async writes drain at S_ENDPGM (implicit idle).

  // ---- epilogue: C/D layout VGPR v -> M = v (+8 for lane half) ----
  const int mbase = m0 + mw * 32 + lhi * 8;
  const int nbase = n0 + nw * 64 + lm;
#pragma unroll
  for (int bn = 0; bn < 4; ++bn) {
    int n = nbase + bn * 16;
    if (n >= N) continue;                     // only tail of the final GEMM
#pragma unroll
    for (int am = 0; am < 2; ++am) {
      int mr = mbase + am * 16;
#pragma unroll
      for (int v = 0; v < 8; ++v) {
        int m = mr + v;                       // M always multiple of 128
        float val = alpha * acc[am][bn][v];
        size_t off = (size_t)m * ldc + n;
        if (Rp) val += Rp[off];
        if (relu) val = fmaxf(val, 0.0f);
        if (Fp) Fp[off] = val;
        if (Bp) Bp[off] = (bf16_t)val;
      }
    }
  }
}

// ---------------------------------------------------------------------------
// Row softmax (f32 -> bf16), one block per row, blockDim.x threads (<=256)
// ---------------------------------------------------------------------------
__global__ void softmax_rows(const float* __restrict__ in,
                             bf16_t* __restrict__ out, int L)
{
  const size_t row = blockIdx.x;
  const float* x = in + row * (size_t)L;
  bf16_t* y = out + row * (size_t)L;
  __shared__ float red[256];
  const int t = threadIdx.x, T = blockDim.x;

  float mx = -__builtin_inff();
  for (int i = t; i < L; i += T) mx = fmaxf(mx, x[i]);
  red[t] = mx; __syncthreads();
  for (int s = T >> 1; s > 0; s >>= 1) {
    if (t < s) red[t] = fmaxf(red[t], red[t + s]);
    __syncthreads();
  }
  mx = red[0]; __syncthreads();

  float sum = 0.0f;
  for (int i = t; i < L; i += T) sum += __expf(x[i] - mx);
  red[t] = sum; __syncthreads();
  for (int s = T >> 1; s > 0; s >>= 1) {
    if (t < s) red[t] += red[t + s];
    __syncthreads();
  }
  const float inv = 1.0f / red[0];
  for (int i = t; i < L; i += T) y[i] = (bf16_t)(__expf(x[i] - mx) * inv);
}

// ---------------------------------------------------------------------------
// Tokenizer softmax: input S (HWP x 128) per batch, softmax over h (rows,
// h<HWL) for each token column t; result written TRANSPOSED as (128 x HWP)
// so the following GEMM consumes it row-major along K=h.
// Block: (128 tokens, 8 h-slices) per batch -> coalesced reads.
// ---------------------------------------------------------------------------
__global__ void softmax_cols_T(const float* __restrict__ in,
                               bf16_t* __restrict__ outT)
{
  const float* S = in + (size_t)blockIdx.x * HWP * NTOK;
  bf16_t* O = outT + (size_t)blockIdx.x * NTOK * HWP;
  __shared__ float red[8][NTOK];
  __shared__ float mrow[NTOK], srow[NTOK];
  const int t = threadIdx.x, s = threadIdx.y;

  float mx = -__builtin_inff();
  for (int h = s; h < HWL; h += 8) mx = fmaxf(mx, S[(size_t)h * NTOK + t]);
  red[s][t] = mx; __syncthreads();
  if (s == 0) {
#pragma unroll
    for (int j = 1; j < 8; ++j) mx = fmaxf(mx, red[j][t]);
    mrow[t] = mx;
  }
  __syncthreads();
  mx = mrow[t];

  float sum = 0.0f;
  for (int h = s; h < HWL; h += 8) sum += __expf(S[(size_t)h * NTOK + t] - mx);
  __syncthreads();
  red[s][t] = sum; __syncthreads();
  if (s == 0) {
    sum = red[0][t];
#pragma unroll
    for (int j = 1; j < 8; ++j) sum += red[j][t];
    srow[t] = sum;
  }
  __syncthreads();
  const float inv = 1.0f / srow[t];
  for (int h = s; h < HWL; h += 8)
    O[(size_t)t * HWP + h] = (bf16_t)(__expf(S[(size_t)h * NTOK + t] - mx) * inv);
}

// ---------------------------------------------------------------------------
// LayerNorm over the whole (CTOK x NTOK) slab per batch; data is TOKEN-major
// (t, c) so affine (c, t) index = (i&1023)<<7 | (i>>10). Optional outputs:
// token-major f32, token-major bf16, channel-major f32 (for d_out T region).
// ---------------------------------------------------------------------------
__global__ __launch_bounds__(256) void layernorm_tok(
    const float* __restrict__ in, const float* __restrict__ w,
    const float* __restrict__ b, float* __restrict__ outTokF,
    bf16_t* __restrict__ outTokB, float* __restrict__ outChF)
{
  const size_t base = (size_t)blockIdx.x * TOKCN;
  const float* x = in + base;
  float*  of = outTokF ? outTokF + base : nullptr;
  bf16_t* ob = outTokB ? outTokB + base : nullptr;
  float*  oc = outChF  ? outChF  + base : nullptr;
  __shared__ float rs[256], rq[256];
  const int t = threadIdx.x;

  float s = 0.0f, q = 0.0f;
  for (int i = t; i < TOKCN; i += 256) { float v = x[i]; s += v; q += v * v; }
  rs[t] = s; rq[t] = q; __syncthreads();
  for (int st = 128; st > 0; st >>= 1) {
    if (t < st) { rs[t] += rs[t + st]; rq[t] += rq[t + st]; }
    __syncthreads();
  }
  const float mean = rs[0] / (float)TOKCN;
  const float var  = rq[0] / (float)TOKCN - mean * mean;
  const float inv  = rsqrtf(var + 1e-5f);

  for (int i = t; i < TOKCN; i += 256) {
    int wi = ((i & 1023) << 7) | (i >> 10);         // c*128 + t
    float v = (x[i] - mean) * inv * w[wi] + b[wi];
    if (of) of[i]  = v;
    if (ob) ob[i]  = (bf16_t)v;
    if (oc) oc[wi] = v;                             // channel-major scatter
  }
}

// ---------------------------------------------------------------------------
// f32 -> bf16 convert (weights) and transpose-convert (Wq_p^T)
// ---------------------------------------------------------------------------
__global__ __launch_bounds__(256) void cvt_f32_bf16(
    const float* __restrict__ in, bf16_t* __restrict__ out, size_t n)
{
  size_t i = (size_t)blockIdx.x * blockDim.x + threadIdx.x;
  size_t stride = (size_t)gridDim.x * blockDim.x;
  for (; i < n; i += stride) out[i] = (bf16_t)in[i];
}

__global__ __launch_bounds__(256) void tcvt_f32_bf16(
    const float* __restrict__ in, bf16_t* __restrict__ out, int R, int C)
{
  int idx = blockIdx.x * blockDim.x + threadIdx.x;
  if (idx < R * C) {
    int r = idx / C, c = idx % C;
    out[(size_t)c * R + r] = (bf16_t)in[(size_t)r * C + c];
  }
}

// ---------------------------------------------------------------------------
// X converter: reads f32 X (CFM x HWL) per batch, writes bf16 Xb (CFM x HWP,
// zero-padded cols) and bf16 XT (HWP x CFM, zero-padded rows) via LDS tiles.
// Grid (HWP/32, CFM/32, NB), block (32, 8).
// ---------------------------------------------------------------------------
__global__ __launch_bounds__(256) void x_cvt_transpose(
    const float* __restrict__ X, bf16_t* __restrict__ Xb,
    bf16_t* __restrict__ XT)
{
  __shared__ float tile[32][33];
  const int batch = blockIdx.z;
  const float* Xp = X + (size_t)batch * CFM * HWL;
  bf16_t* Xbp = Xb + (size_t)batch * CFM * HWP;
  bf16_t* XTp = XT + (size_t)batch * HWP * CFM;
  const int h0 = blockIdx.x * 32, c0 = blockIdx.y * 32;
  const int tx = threadIdx.x, ty = threadIdx.y;

#pragma unroll
  for (int r = ty; r < 32; r += 8) {
    int c = c0 + r, h = h0 + tx;
    float v = (h < HWL) ? Xp[(size_t)c * HWL + h] : 0.0f;
    tile[r][tx] = v;
    Xbp[(size_t)c * HWP + h] = (bf16_t)v;           // coalesced along h
  }
  __syncthreads();
#pragma unroll
  for (int r = ty; r < 32; r += 8) {
    int h = h0 + r, c = c0 + tx;
    XTp[(size_t)h * CFM + c] = (bf16_t)tile[tx][r]; // coalesced along c
  }
}

// ---------------------------------------------------------------------------
// Host orchestration
// ---------------------------------------------------------------------------
static inline void gemm(hipStream_t st,
    const bf16_t* A, long long sA, int lda,
    const bf16_t* Bm, long long sB, int ldb,
    float alpha, const float* R, long long sR,
    float* F, long long sF, bf16_t* OB, long long sOB,
    int M, int N, int K, int ldc, int relu)
{
  dim3 g((N + 127) / 128, M / 128, NB);
  gemm_nt_wmma<<<g, dim3(256), GEMM_LDS_BYTES, st>>>(
      A, sA, lda, Bm, sB, ldb, alpha, R, sR, F, sF, OB, sOB,
      M, N, K, ldc, relu);
}

static inline void cvt(hipStream_t st, const float* in, bf16_t* out, size_t n)
{
  unsigned blocks = (unsigned)(((n + 255) / 256) > 2048 ? 2048 : ((n + 255) / 256));
  cvt_f32_bf16<<<blocks, 256, 0, st>>>(in, out, n);
}

extern "C" void kernel_launch(void* const* d_in, const int* in_sizes, int n_in,
                              void* d_out, int out_size, void* d_ws, size_t ws_size,
                              hipStream_t stream)
{
  (void)in_sizes; (void)n_in; (void)out_size;
  const float* X      = (const float*)d_in[0];
  const float* W_a    = (const float*)d_in[1];
  const float* W_vtok = (const float*)d_in[2];
  const float* Wk_t   = (const float*)d_in[3];
  const float* Wq_t   = (const float*)d_in[4];
  const float* Wv_t   = (const float*)d_in[5];
  const float* f1     = (const float*)d_in[6];
  const float* f2     = (const float*)d_in[7];
  const float* ln_w   = (const float*)d_in[8];
  const float* ln_b   = (const float*)d_in[9];
  const float* Wq_p   = (const float*)d_in[10];
  const float* Wk_p   = (const float*)d_in[11];
  const float* Wv_p   = (const float*)d_in[12];

  const size_t XOUT = (size_t)NB * CFM * HWL;          // out region size
  const long long sXf = (long long)CFM * HWL;          // f32 X per-batch
  const long long sXb = (long long)CFM * HWP;          // padded bf16 X
  const long long sSc = (long long)HWP * NTOK;         // padded score slab
  const long long sT  = TOKCN;                         // token-tensor stride

  char* ws = (char*)d_ws;
  size_t cur = 0;
  auto alloc = [&](size_t bytes) -> char* {
    char* p = ws + cur;
    cur += (bytes + 255) & ~(size_t)255;
    return p;
  };
  bf16_t* Xb    = (bf16_t*)alloc((size_t)NB * sXb * 2);
  bf16_t* XTb   = (bf16_t*)alloc((size_t)NB * sXb * 2);
  bf16_t* Wab   = (bf16_t*)alloc((size_t)NTOK * CFM * 2);
  bf16_t* Wvt0b = (bf16_t*)alloc((size_t)CTOK * CFM * 2);
  bf16_t* Wktb  = (bf16_t*)alloc((size_t)CTOK * CTOK * 2);
  bf16_t* Wqtb  = (bf16_t*)alloc((size_t)CTOK * CTOK * 2);
  bf16_t* Wvtb  = (bf16_t*)alloc((size_t)CTOK * CTOK * 2);
  bf16_t* f1b   = (bf16_t*)alloc((size_t)CTOK * CTOK * 2);
  bf16_t* f2b   = (bf16_t*)alloc((size_t)CTOK * CTOK * 2);
  bf16_t* Wkpb  = (bf16_t*)alloc((size_t)CFM * CTOK * 2);
  bf16_t* Wvpb  = (bf16_t*)alloc((size_t)CFM * CTOK * 2);
  bf16_t* WqpTb = (bf16_t*)alloc((size_t)CFM * CFM * 2);
  float*  Sc    = (float*) alloc((size_t)NB * sSc * 4);  // ScHT then Sp
  bf16_t* Attn  = (bf16_t*)alloc((size_t)NB * sSc * 2);  // attn_th then attn_p
  bf16_t* m1Tb  = (bf16_t*)alloc((size_t)NB * sT * 2);
  float*  tTf   = (float*) alloc((size_t)NB * sT * 4);
  bf16_t* tTb   = (bf16_t*)alloc((size_t)NB * sT * 2);
  bf16_t* kTb   = (bf16_t*)alloc((size_t)NB * sT * 2);
  bf16_t* qTb   = (bf16_t*)alloc((size_t)NB * sT * 2);
  bf16_t* vCMb  = (bf16_t*)alloc((size_t)NB * sT * 2);
  float*  S2f   = (float*) alloc((size_t)NB * NTOK * NTOK * 4);
  bf16_t* attnJ = (bf16_t*)alloc((size_t)NB * NTOK * NTOK * 2);
  float*  Pre   = (float*) alloc((size_t)NB * sT * 4);   // T1pre, reused as T2pre
  float*  t1Tf  = (float*) alloc((size_t)NB * sT * 4);
  bf16_t* t1Tb  = (bf16_t*)alloc((size_t)NB * sT * 2);
  bf16_t* hTb   = (bf16_t*)alloc((size_t)NB * sT * 2);
  bf16_t* toTb  = (bf16_t*)alloc((size_t)NB * sT * 2);
  bf16_t* kpTb  = (bf16_t*)alloc((size_t)NB * sT * 2);
  bf16_t* kp2Tb = (bf16_t*)alloc((size_t)NB * sT * 2);
  bf16_t* vpCMb = (bf16_t*)alloc((size_t)NB * sT * 2);
  if (cur > ws_size) return;

  float* outX = (float*)d_out;           // (B, CFM, HWL)
  float* outT = (float*)d_out + XOUT;    // (B, CTOK, NTOK) channel-major
  const float rsc = 0.03125f;            // 1/sqrt(1024)

  // ---- precision conversion + X transpose (pads HW to 3200 with zeros) ----
  x_cvt_transpose<<<dim3(HWP / 32, CFM / 32, NB), dim3(32, 8), 0, stream>>>(X, Xb, XTb);
  cvt(stream, W_a,    Wab,   (size_t)NTOK * CFM);
  cvt(stream, W_vtok, Wvt0b, (size_t)CTOK * CFM);
  cvt(stream, Wk_t,   Wktb,  (size_t)CTOK * CTOK);
  cvt(stream, Wq_t,   Wqtb,  (size_t)CTOK * CTOK);
  cvt(stream, Wv_t,   Wvtb,  (size_t)CTOK * CTOK);
  cvt(stream, f1,     f1b,   (size_t)CTOK * CTOK);
  cvt(stream, f2,     f2b,   (size_t)CTOK * CTOK);
  cvt(stream, Wk_p,   Wkpb,  (size_t)CFM * CTOK);
  cvt(stream, Wv_p,   Wvpb,  (size_t)CFM * CTOK);
  tcvt_f32_bf16<<<(CFM * CFM + 255) / 256, 256, 0, stream>>>(Wq_p, WqpTb, CFM, CFM);

  // ---- FilterBasedTokenizer ----
  // ScHT[h,t] = X^T[h,:] . W_a[t,:] / 32        (3200 x 128)
  gemm(stream, XTb, sXb, CFM, Wab, 0, CFM, rsc,
       nullptr, 0, Sc, sSc, nullptr, 0, HWP, NTOK, CFM, NTOK, 0);
  // softmax over h per token; write transposed attn_th (128 x 3200)
  softmax_cols_T<<<NB, dim3(NTOK, 8), 0, stream>>>(Sc, Attn);
  // m1T[t,c] = attn_th[t,:] . Xb[c,:]           (128 x 1024), K = 3136
  gemm(stream, Attn, sSc, HWP, Xb, sXb, HWP, 1.0f,
       nullptr, 0, nullptr, 0, m1Tb, sT, NTOK, CFM, HWL, CFM, 0);
  // tT[t,c'] = m1T[t,:] . W_vtok[c',:]  == T^T  (token-major T)
  gemm(stream, m1Tb, sT, CFM, Wvt0b, 0, CFM, 1.0f,
       nullptr, 0, tTf, sT, tTb, sT, NTOK, CTOK, CFM, CTOK, 0);

  // ---- Transformer ----
  gemm(stream, tTb, sT, CTOK, Wktb, 0, CTOK, 1.0f,
       nullptr, 0, nullptr, 0, kTb, sT, NTOK, CTOK, CTOK, CTOK, 0);
  gemm(stream, tTb, sT, CTOK, Wqtb, 0, CTOK, 1.0f,
       nullptr, 0, nullptr, 0, qTb, sT, NTOK, CTOK, CTOK, CTOK, 0);
  // V channel-major: vCM[c,i] = Wv_t[c,:] . tT[i,:]
  gemm(stream, Wvtb, 0, CTOK, tTb, sT, CTOK, 1.0f,
       nullptr, 0, nullptr, 0, vCMb, sT, CTOK, NTOK, CTOK, NTOK, 0);
  // S2[j,i] = qT[j,:] . kT[i,:] / 32 = scores[i,j]; row softmax == axis=1
  gemm(stream, qTb, sT, CTOK, kTb, sT, CTOK, rsc,
       nullptr, 0, S2f, (long long)NTOK * NTOK, nullptr, 0,
       NTOK, NTOK, CTOK, NTOK, 0);
  softmax_rows<<<NB * NTOK, 128, 0, stream>>>(S2f, attnJ, NTOK);
  // aoT[j,c] = attnJ[j,:] . vCM[c,:] + tT  ->  T1pre (token-major)
  gemm(stream, attnJ, (long long)NTOK * NTOK, NTOK, vCMb, sT, NTOK, 1.0f,
       tTf, sT, Pre, sT, nullptr, 0, NTOK, CTOK, NTOK, CTOK, 0);
  layernorm_tok<<<NB, 256, 0, stream>>>(Pre, ln_w, ln_b, t1Tf, t1Tb, nullptr);
  // FFN
  gemm(stream, t1Tb, sT, CTOK, f1b, 0, CTOK, 1.0f,
       nullptr, 0, nullptr, 0, hTb, sT, NTOK, CTOK, CTOK, CTOK, 1);
  gemm(stream, hTb, sT, CTOK, f2b, 0, CTOK, 1.0f,
       t1Tf, sT, Pre, sT, nullptr, 0, NTOK, CTOK, CTOK, CTOK, 0);
  // T_out: bf16 token-major for projector + f32 channel-major into d_out
  layernorm_tok<<<NB, 256, 0, stream>>>(Pre, ln_w, ln_b, nullptr, toTb, outT);

  // ---- Projector ----
  // kpT[l,c] = toT[l,:] . Wk_p[c,:]      (Kp token-major)
  gemm(stream, toTb, sT, CTOK, Wkpb, 0, CTOK, 1.0f,
       nullptr, 0, nullptr, 0, kpTb, sT, NTOK, CFM, CTOK, CFM, 0);
  // kp2T[l,c] = kpT[l,:] . WqpT[c,:]     (Wq_p^T Kp, token-major)
  gemm(stream, kpTb, sT, CFM, WqpTb, 0, CFM, 1.0f,
       nullptr, 0, nullptr, 0, kp2Tb, sT, NTOK, CFM, CFM, CFM, 0);
  // Sp[h,l] = XT[h,:] . kp2T[l,:] / 32   (3200 x 128)
  gemm(stream, XTb, sXb, CFM, kp2Tb, sT, CFM, rsc,
       nullptr, 0, Sc, sSc, nullptr, 0, HWP, NTOK, CFM, NTOK, 0);
  softmax_rows<<<NB * HWP, 128, 0, stream>>>(Sc, Attn, NTOK);
  // Vp channel-major: vpCM[c,l] = Wv_p[c,:] . toT[l,:]
  gemm(stream, Wvpb, 0, CTOK, toTb, sT, CTOK, 1.0f,
       nullptr, 0, nullptr, 0, vpCMb, sT, CFM, NTOK, CTOK, NTOK, 0);
  // out[c,h] = X[c,h] + vpCM[c,:] . attn_p[h,:]   (N tail guarded)
  gemm(stream, vpCMb, sT, NTOK, Attn, sSc, NTOK, 1.0f,
       X, sXf, outX, sXf, nullptr, 0, CFM, HWL, NTOK, HWL, 0);
}
